// MambaLayer_45698452029746
// MI455X (gfx1250) — matrix operand order repaired and checked by reading, not compiled, not run on hardware
//
#include <hip/hip_runtime.h>
#include <hip/hip_bf16.h>

// ---------------------------------------------------------------------------
// Mamba layer forward for MI455X (gfx1250, wave32, WMMA).
// B=2, L=2048, D_MODEL=1024, D_INNER=2048, D_STATE=16, D_CONV=4, DT_RANK=64
// bf16 WMMA (v_wmma_f32_16x16x32_bf16) for all four GEMMs; fused pointwise
// kernels for conv/softplus/scan/gating. GEMMs are bandwidth-bound on
// MI455X (23.3 TB/s) once on the WMMA path, so every intermediate is
// read/written exactly once.
// ---------------------------------------------------------------------------

#define B_SZ     2
#define L_SZ     2048
#define D_MODEL  1024
#define D_INNER  2048
#define D_STATE  16
#define D_CONV   4
#define DT_RANK  64
#define M_ROWS   (B_SZ * L_SZ)           // 4096 token rows (multiple of 128)
#define N_XZ     (2 * D_INNER)           // 4096
#define N_XPROJ  (DT_RANK + 2 * D_STATE) // 96

typedef __attribute__((ext_vector_type(16))) __bf16 v16bf;
typedef __attribute__((ext_vector_type(8)))  __bf16 v8bf;
typedef __attribute__((ext_vector_type(4)))  __bf16 v4bf;
typedef __attribute__((ext_vector_type(8)))  float  v8f;
typedef __attribute__((ext_vector_type(4)))  float  v4f;

__device__ __forceinline__ __bf16 to_bf16(float f) {
    unsigned int u = __builtin_bit_cast(unsigned int, f);
    unsigned int r = u + 0x7FFFu + ((u >> 16) & 1u);   // round-to-nearest-even
    unsigned short h = (unsigned short)(r >> 16);
    return __builtin_bit_cast(__bf16, h);
}

// ---------------------------------------------------------------------------
// fp32 -> bf16 conversion, 4 elements/thread (16B in, 8B out per thread).
// All call sites have n % 4 == 0.
// ---------------------------------------------------------------------------
__global__ __launch_bounds__(256) void k_convert_bf16_v4(
        const float* __restrict__ src, __bf16* __restrict__ dst, int n4) {
    int i = blockIdx.x * blockDim.x + threadIdx.x;
    int stride = gridDim.x * blockDim.x;
    for (; i < n4; i += stride) {
        v4f f = *(const v4f*)(src + 4 * (size_t)i);
        v4bf o;
#pragma unroll
        for (int e = 0; e < 4; ++e) o[e] = to_bf16(f[e]);
        *(v4bf*)(dst + 4 * (size_t)i) = o;
    }
}

// ---------------------------------------------------------------------------
// bf16 WMMA GEMM:  C[M,N] = A[M,K] * W[N,K]^T   (f32 accumulate)
// Block: 256 threads = 8 waves -> 128x128 tile.
// Wave: 32x64 tile = 2x4 accumulators of v_wmma_f32_16x16x32_bf16.
// M is always a multiple of 128 (no row guard). NGUARD: col guard for N=96.
// EPI==0: plain store.  EPI==1: softplus(acc + bias[col]).
// ---------------------------------------------------------------------------
template <int EPI, bool NGUARD>
__global__ __launch_bounds__(256) void k_gemm_bf16_wmma(
        const __bf16* __restrict__ A, const __bf16* __restrict__ W,
        float* __restrict__ C, int M, int N, int K,
        const float* __restrict__ bias) {
    const int tid  = threadIdx.x;
    const int lane = tid & 31;
    const int wid  = tid >> 5;
    const int lc   = lane & 15;      // row/col within a 16-tile
    const int hi   = lane >> 4;      // which 16-lane half

    const int m0 = blockIdx.x * 128 + (wid & 3) * 32;   // wave M origin
    const int n0 = blockIdx.y * 128 + (wid >> 2) * 64;  // wave N origin

    v8f acc[2][4];
#pragma unroll
    for (int i = 0; i < 2; ++i)
#pragma unroll
        for (int j = 0; j < 4; ++j) acc[i][j] = (v8f)(0.0f);

    // ---- per-lane 32-bit element offsets (all buffers < 64 MB) ----
    // A fragment (ISA 16-bit A layout): lane holds row m0+i*16+lc,
    //   K-halves hi*8 .. +7 and hi*8+16 .. +23.
    unsigned aoff[2];
#pragma unroll
    for (int i = 0; i < 2; ++i)
        aoff[i] = (unsigned)(m0 + i * 16 + lc) * (unsigned)K + (unsigned)(hi * 8);
    // B fragment: same K order as A: K-halves hi*8 .. +7 and hi*8+16 .. +23.
    unsigned boff[4];
#pragma unroll
    for (int j = 0; j < 4; ++j) {
        int col = n0 + j * 16 + lc;
        if (NGUARD) { if (col > N - 1) col = N - 1; }
        boff[j] = (unsigned)col * (unsigned)K + (unsigned)(hi * 8);
    }

    for (int k0 = 0; k0 < K; k0 += 32) {
        v16bf afrag[2];
#pragma unroll
        for (int i = 0; i < 2; ++i) {
            const __bf16* pa = A + aoff[i] + k0;
            v8bf lo = *(const v8bf*)(pa);        // 16B
            v8bf hp = *(const v8bf*)(pa + 16);   // 16B
#pragma unroll
            for (int e = 0; e < 8; ++e) { afrag[i][e] = lo[e]; afrag[i][8 + e] = hp[e]; }
        }
        v16bf bfrag[4];
#pragma unroll
        for (int j = 0; j < 4; ++j) {
            const __bf16* pb = W + boff[j] + k0;
            v8bf blo = *(const v8bf*)(pb);        // 16B
            v8bf bhp = *(const v8bf*)(pb + 16);   // 16B
#pragma unroll
            for (int e = 0; e < 8; ++e) { bfrag[j][e] = blo[e]; bfrag[j][8 + e] = bhp[e]; }
        }
#pragma unroll
        for (int i = 0; i < 2; ++i)
#pragma unroll
            for (int j = 0; j < 4; ++j)
                acc[i][j] = __builtin_amdgcn_wmma_f32_16x16x32_bf16(
                    false, afrag[i], false, bfrag[j],
                    (short)0, acc[i][j], false, false);
    }

    // ---- store: C/D layout = VGPR r -> row r + 8*hi, col = lane%16 ----
    // Straight-line stores, 32-bit offsets strength-reduced by +N per row.
#pragma unroll
    for (int i = 0; i < 2; ++i) {
        const unsigned rbase = (unsigned)(m0 + i * 16 + hi * 8);
#pragma unroll
        for (int j = 0; j < 4; ++j) {
            const int col = n0 + j * 16 + lc;
            if (NGUARD && col >= N) continue;
            float bv = 0.0f;
            if (EPI == 1) bv = bias[col];
            unsigned off = rbase * (unsigned)N + (unsigned)col;
#pragma unroll
            for (int r = 0; r < 8; ++r) {
                float v = acc[i][j][r];
                if (EPI == 1) {  // softplus(v + b_dt[col])
                    v += bv;
                    v = (v > 20.0f) ? v : log1pf(__expf(v));
                }
                C[off] = v;
                off += (unsigned)N;
            }
        }
    }
}

// ---------------------------------------------------------------------------
// Depthwise causal conv(4) + bias + SiLU.  Reads xi = xz[:, :D_INNER].
// Writes xc (f32, for the scan) and xc_bf (bf16, for the x-proj GEMM).
// ---------------------------------------------------------------------------
__global__ __launch_bounds__(256) void k_conv_silu(
        const float* __restrict__ xz, const float* __restrict__ conv_w,
        const float* __restrict__ conv_b, float* __restrict__ xc,
        __bf16* __restrict__ xc_bf) {
    int idx = blockIdx.x * blockDim.x + threadIdx.x;
    const int d   = idx & (D_INNER - 1);
    const int row = idx >> 11;          // D_INNER == 2048
    const int t   = row & (L_SZ - 1);
    float a = conv_b[d];
#pragma unroll
    for (int k = 0; k < D_CONV; ++k) {
        const int tt = t - (D_CONV - 1) + k;
        if (tt >= 0)
            a += conv_w[d * D_CONV + k] *
                 xz[(size_t)(row - (D_CONV - 1) + k) * N_XZ + d];
    }
    const float v = a / (1.0f + __expf(-a));   // SiLU
    xc[idx]    = v;
    xc_bf[idx] = to_bf16(v);
}

// ---------------------------------------------------------------------------
// Extract dt_r (first 64 cols of x_dbl) as bf16 for the dt GEMM.
// ---------------------------------------------------------------------------
__global__ __launch_bounds__(256) void k_extract_dtr(
        const float* __restrict__ x_dbl, __bf16* __restrict__ dtr_bf) {
    int idx = blockIdx.x * blockDim.x + threadIdx.x;
    const int c   = idx & (DT_RANK - 1);
    const int row = idx >> 6;
    dtr_bf[idx] = to_bf16(x_dbl[(size_t)row * N_XPROJ + c]);
}

// ---------------------------------------------------------------------------
// Selective scan. One thread per (b,d) channel; h[16] in registers;
// per-timestep B/C vectors staged in LDS by lanes 0..31.
// Fused: y = (sum_n h*C + Dp*u) * SiLU(z), emitted as bf16 for out-proj GEMM.
// ---------------------------------------------------------------------------
__global__ __launch_bounds__(256) void k_scan(
        const float* __restrict__ dt, const float* __restrict__ xc,
        const float* __restrict__ x_dbl, const float* __restrict__ xz,
        const float* __restrict__ A_log, const float* __restrict__ Dp,
        __bf16* __restrict__ y_bf) {
    const int b = blockIdx.y;
    const int d = blockIdx.x * blockDim.x + threadIdx.x;   // 0..D_INNER-1

    float Av[D_STATE];
#pragma unroll
    for (int n = 0; n < D_STATE; ++n) Av[n] = -__expf(A_log[d * D_STATE + n]);
    float h[D_STATE];
#pragma unroll
    for (int n = 0; n < D_STATE; ++n) h[n] = 0.0f;
    const float Dv = Dp[d];

    __shared__ float sBC[2 * D_STATE];   // [0:16]=B_t, [16:32]=C_t

    for (int t = 0; t < L_SZ; ++t) {
        const size_t row = (size_t)b * L_SZ + t;
        __syncthreads();
        if (threadIdx.x < 2 * D_STATE)
            sBC[threadIdx.x] = x_dbl[row * N_XPROJ + DT_RANK + threadIdx.x];
        __syncthreads();

        const float dtv = dt[row * D_INNER + d];
        const float u   = xc[row * D_INNER + d];
        const float zv  = xz[row * N_XZ + D_INNER + d];
        const float du  = dtv * u;
        float yv = 0.0f;
#pragma unroll
        for (int n = 0; n < D_STATE; ++n) {
            const float dA = __expf(dtv * Av[n]);
            h[n] = dA * h[n] + du * sBC[n];
            yv += h[n] * sBC[D_STATE + n];
        }
        yv += Dv * u;
        yv *= zv / (1.0f + __expf(-zv));           // * SiLU(z)
        y_bf[row * D_INNER + d] = to_bf16(yv);
    }
}

// ---------------------------------------------------------------------------
// Host-side launch
// ---------------------------------------------------------------------------
static inline size_t align256(size_t x) { return (x + 255) & ~(size_t)255; }

extern "C" void kernel_launch(void* const* d_in, const int* in_sizes, int n_in,
                              void* d_out, int out_size, void* d_ws, size_t ws_size,
                              hipStream_t stream) {
    (void)in_sizes; (void)n_in; (void)out_size; (void)ws_size;
    const float* x      = (const float*)d_in[0];
    const float* W_in   = (const float*)d_in[1];
    const float* conv_w = (const float*)d_in[2];
    const float* conv_b = (const float*)d_in[3];
    const float* W_xprj = (const float*)d_in[4];
    const float* W_dt   = (const float*)d_in[5];
    const float* b_dt   = (const float*)d_in[6];
    const float* A_log  = (const float*)d_in[7];
    const float* Dp     = (const float*)d_in[8];
    const float* W_out  = (const float*)d_in[9];
    float* out = (float*)d_out;

    // ---- workspace layout (bytes), total ~192 MB ----
    char* ws = (char*)d_ws;
    size_t o = 0;
    float*  xz     = (float*)(ws + o); o = align256(o + (size_t)M_ROWS * N_XZ   * 4); // 64 MB
    float*  xc     = (float*)(ws + o); o = align256(o + (size_t)M_ROWS * D_INNER* 4); // 32 MB
    float*  dt     = (float*)(ws + o); o = align256(o + (size_t)M_ROWS * D_INNER* 4); // 32 MB
    float*  x_dbl  = (float*)(ws + o); o = align256(o + (size_t)M_ROWS * N_XPROJ* 4); // 1.5 MB
    __bf16* x_bf   = (__bf16*)(ws + o); o = align256(o + (size_t)M_ROWS * D_MODEL * 2);
    __bf16* win_bf = (__bf16*)(ws + o); o = align256(o + (size_t)N_XZ   * D_MODEL * 2);
    __bf16* xc_bf  = (__bf16*)(ws + o); o = align256(o + (size_t)M_ROWS * D_INNER * 2);
    __bf16* wxp_bf = (__bf16*)(ws + o); o = align256(o + (size_t)N_XPROJ* D_INNER * 2);
    __bf16* dtr_bf = (__bf16*)(ws + o); o = align256(o + (size_t)M_ROWS * DT_RANK * 2);
    __bf16* wdt_bf = (__bf16*)(ws + o); o = align256(o + (size_t)D_INNER* DT_RANK * 2);
    __bf16* y_bf   = (__bf16*)(ws + o); o = align256(o + (size_t)M_ROWS * D_INNER * 2);
    __bf16* wout_bf= (__bf16*)(ws + o); o = align256(o + (size_t)D_MODEL* D_INNER * 2);

    const int CT = 256;
    auto cvt = [&](const float* s, __bf16* d, int n) {
        int n4 = n / 4;
        int blocks = (n4 + CT - 1) / CT; if (blocks > 8192) blocks = 8192;
        k_convert_bf16_v4<<<blocks, CT, 0, stream>>>(s, d, n4);
    };
    // 1) bf16 conversions
    cvt(x,      x_bf,    M_ROWS * D_MODEL);
    cvt(W_in,   win_bf,  N_XZ * D_MODEL);
    cvt(W_xprj, wxp_bf,  N_XPROJ * D_INNER);
    cvt(W_dt,   wdt_bf,  D_INNER * DT_RANK);
    cvt(W_out,  wout_bf, D_MODEL * D_INNER);

    // 2) in-projection:  xz[M,4096] = x_bf[M,1024] * W_in^T
    k_gemm_bf16_wmma<0, false><<<dim3(M_ROWS / 128, N_XZ / 128), 256, 0, stream>>>(
        x_bf, win_bf, xz, M_ROWS, N_XZ, D_MODEL, nullptr);

    // 3) depthwise conv + SiLU -> xc (f32 + bf16)
    k_conv_silu<<<(M_ROWS * D_INNER) / CT, CT, 0, stream>>>(
        xz, conv_w, conv_b, xc, xc_bf);

    // 4) x-projection:  x_dbl[M,96] = xc_bf * W_xproj^T   (N=96 -> guarded)
    k_gemm_bf16_wmma<0, true><<<dim3(M_ROWS / 128, (N_XPROJ + 127) / 128), 256, 0, stream>>>(
        xc_bf, wxp_bf, x_dbl, M_ROWS, N_XPROJ, D_INNER, nullptr);

    // 5) dt_r -> bf16
    k_extract_dtr<<<(M_ROWS * DT_RANK) / CT, CT, 0, stream>>>(x_dbl, dtr_bf);

    // 6) dt-projection + fused softplus(+b_dt):  dt[M,2048]
    k_gemm_bf16_wmma<1, false><<<dim3(M_ROWS / 128, D_INNER / 128), 256, 0, stream>>>(
        dtr_bf, wdt_bf, dt, M_ROWS, D_INNER, DT_RANK, b_dt);

    // 7) selective scan + D-skip + SiLU(z) gating -> y_bf
    k_scan<<<dim3(D_INNER / CT, B_SZ), CT, 0, stream>>>(
        dt, xc, x_dbl, xz, A_log, Dp, y_bf);

    // 8) out-projection straight into d_out: out[M,1024] = y_bf * W_out^T
    k_gemm_bf16_wmma<0, false><<<dim3(M_ROWS / 128, D_MODEL / 128), 256, 0, stream>>>(
        y_bf, wout_bf, out, M_ROWS, D_MODEL, D_INNER, nullptr);
}
